// Nystrom_attention_partial_structure_12103217840235
// MI455X (gfx1250) — matrix operand order repaired and checked by reading, not compiled, not run on hardware
//
#include <hip/hip_runtime.h>

// ---------------------------------------------------------------------------
// CDNA5 (gfx1250, wave32) Nystrom attention w/ partial structure.
//  - all GEMMs: v_wmma_f32_16x16x32_bf16, f32 accumulation
//  - operands converted to bf16 once in producer kernels
//  - GEMM tiles staged into LDS with the Tensor Data Mover
//    (tensor_load_to_lds + s_wait_tensorcnt), double-buffered
//  - big GEMMs (QKV, final projection): 128x128 block tiles, 8 waves
// ---------------------------------------------------------------------------

typedef __bf16 bf16;
typedef bf16  v16bf __attribute__((ext_vector_type(16)));
typedef float v8f   __attribute__((ext_vector_type(8)));
typedef unsigned int u32x4 __attribute__((ext_vector_type(4)));
typedef int          i32x4 __attribute__((ext_vector_type(4)));
typedef int          i32x8 __attribute__((ext_vector_type(8)));

#define LDK    72   // LDS ushort stride, 64-wide tiles  (144B rows, 16B aligned)
#define LDB128 136  // LDS ushort stride, 128-wide tiles (272B rows, 16B aligned)
#define ATILE128 (128 * LDK)     // ushorts per 128x64 A tile
#define BTILE128 (64 * LDB128)   // ushorts per 64x128 B tile

__device__ __forceinline__ unsigned short f2bf(float x) {
  unsigned u = __builtin_bit_cast(unsigned, x);
  u += 0x7FFFu + ((u >> 16) & 1u);           // round-to-nearest-even
  return (unsigned short)(u >> 16);
}
__device__ __forceinline__ float bf2f(unsigned short u) {
  return __builtin_bit_cast(float, (unsigned)u << 16);
}

union FragBF { v16bf v; unsigned short u[16]; uint4 q[2]; };
union U4x8   { uint4 q; unsigned short s[8]; };

__device__ __forceinline__ v8f wmma_bf16(v16bf a, v16bf b, v8f c) {
  return __builtin_amdgcn_wmma_f32_16x16x32_bf16(false, a, false, b, (short)0, c, false, false);
}

// A fragment (16x32 bf16), LDS row-major, stride ld ushorts.
__device__ __forceinline__ v16bf frag_a(const unsigned short* As, int ld, int row_base, int k0, int lane) {
  int lr = lane & 15, hi = lane >> 4;
  const unsigned short* p = As + (row_base + lr) * ld + k0 + 8 * hi;
  FragBF f;
  f.q[0] = *(const uint4*)p;
  f.q[1] = *(const uint4*)(p + 16);
  return f.v;
}
// B fragment (32x16 bf16), LDS K-major, stride ld. lane%16 = N, K in table order.
__device__ __forceinline__ v16bf frag_b(const unsigned short* Bs, int ld, int k0, int col_base, int lane) {
  int lr = lane & 15, hi = lane >> 4;
  const unsigned short* p = Bs + (k0 + 8 * hi) * ld + col_base + lr;
  FragBF f;
  #pragma unroll
  for (int j = 0; j < 8; ++j) { f.u[j] = p[j * ld]; f.u[8 + j] = p[(16 + j) * ld]; }
  return f.v;
}

// ---------------------------------------------------------------------------
// Tensor Data Mover: async-load a (rows x cols) bf16 tile (row stride `stride`
// elements) into LDS at byte offset lds_off, inserting 16B of pad after each
// row (pad_code: 4 -> after 128B row, 5 -> after 256B row) so the LDS layout
// matches LDK / LDB128.  D# fields per CDNA5 ISA.
// ---------------------------------------------------------------------------
#if defined(__has_builtin)
#if __has_builtin(__builtin_amdgcn_tensor_load_to_lds) && __has_builtin(__builtin_amdgcn_s_wait_tensorcnt)
#define HAS_TDM 1
#endif
#endif
#ifndef HAS_TDM
#define HAS_TDM 0
#endif

#if HAS_TDM
__device__ __forceinline__ void tdm_load_tile(unsigned lds_off, const unsigned short* g, int stride,
                                              int rows, int cols, int pad_code) {
  unsigned long long ga = (unsigned long long)g;
  u32x4 g0;
  g0[0] = 1u;                                                 // count=1 valid descriptor
  g0[1] = lds_off;                                            // lds_addr (bytes)
  g0[2] = (unsigned)(ga & 0xFFFFFFFFull);                     // global_addr[31:0]
  g0[3] = (unsigned)((ga >> 32) & 0x1FFFFFFull) | (2u << 30); // addr[56:32] | type=2
  i32x8 g1;
  // data_size=2B, pad_enable, pad_interval=pad_code, pad_amount=4 DW (16B)
  g1[0] = (1 << 16) | (1 << 20) | (pad_code << 22) | (3 << 25);
  // tensor_dim0 = tensor_dim1 = 1<<20 (tiles are interior; no OOB clipping)
  g1[1] = 0;                         // abar=0, tensor_dim0[15:0]=0
  g1[2] = 0x10;                      // tensor_dim0[31:16]=0x10, tensor_dim1[15:0]=0
  g1[3] = 0x10 | (cols << 16);       // tensor_dim1[31:16]=0x10, tile_dim0=cols
  g1[4] = rows;                      // tile_dim1=rows, tile_dim2=0
  g1[5] = stride;                    // tensor_dim0_stride[31:0] (elements)
  g1[6] = 0;
  g1[7] = 0;
  i32x4 zz = {0, 0, 0, 0};
#if defined(__clang_major__) && (__clang_major__ >= 23)
  i32x8 z8 = {0, 0, 0, 0, 0, 0, 0, 0};
  __builtin_amdgcn_tensor_load_to_lds(g0, g1, zz, zz, z8, 0);
#else
  __builtin_amdgcn_tensor_load_to_lds(g0, g1, zz, zz, 0);
#endif
}
__device__ __forceinline__ unsigned lds_offset(const void* p) {
  return (unsigned)(unsigned long long)(__SIZE_TYPE__)p;      // flat addr[31:0] == LDS offset
}
#endif

// manual fallback / transposed-B staging (bf16 source), 64x64, 128 threads
__device__ __forceinline__ void stage_tile_rowmajor(unsigned short* dst, const unsigned short* src,
                                                    long ld, int tid) {
  #pragma unroll
  for (int it = 0; it < 4; ++it) {
    int idx = tid + it * 128;
    int r = idx >> 3, c8 = (idx & 7) << 3;
    *(uint4*)&dst[r * LDK + c8] = *(const uint4*)(src + (long)r * ld + c8);
  }
}
__device__ __forceinline__ void stage_tile_transposed(unsigned short* dst, const unsigned short* src,
                                                      long ld, int tid) {
  #pragma unroll
  for (int it = 0; it < 4; ++it) {
    int idx = tid + it * 128;
    int nn = idx >> 3, c8 = (idx & 7) << 3;
    U4x8 u; u.q = *(const uint4*)(src + (long)nn * ld + c8);
    #pragma unroll
    for (int j = 0; j < 8; ++j) dst[(c8 + j) * LDK + nn] = u.s[j];
  }
}
// manual fallback, 128-tile variants, 256 threads
__device__ __forceinline__ void stage_a128(unsigned short* dst, const unsigned short* src,
                                           long ld, int tid) {
  #pragma unroll
  for (int it = 0; it < 4; ++it) {
    int idx = tid + it * 256;                 // 1024 x uint4 = 128x64 ushorts
    int r = idx >> 3, c8 = (idx & 7) << 3;
    *(uint4*)&dst[r * LDK + c8] = *(const uint4*)(src + (long)r * ld + c8);
  }
}
__device__ __forceinline__ void stage_b128(unsigned short* dst, const unsigned short* src,
                                           long ld, int tid) {
  #pragma unroll
  for (int it = 0; it < 4; ++it) {
    int idx = tid + it * 256;                 // 1024 x uint4 = 64x128 ushorts
    int r = idx >> 4, c8 = (idx & 15) << 3;
    *(uint4*)&dst[r * LDB128 + c8] = *(const uint4*)(src + (long)r * ld + c8);
  }
}

// ---------------------------------------------------------------------------
// Generic batched GEMM (64x64 tiles, 4 waves): C[z] (+)= A[z] @ B[z] / A @ B^T.
// ---------------------------------------------------------------------------
template <bool TRANSB, bool ACCUM>
__global__ __launch_bounds__(128) void gemm_bf16(
    const unsigned short* __restrict__ A, const unsigned short* __restrict__ B,
    float* __restrict__ C,
    int M, int N, int K, int lda, int ldb, int ldc,
    long sA, long sB, long sC)
{
  __shared__ unsigned short As[2][64 * LDK];
  __shared__ unsigned short Bs[2][64 * LDK];
  const int z = blockIdx.z;
  A += (long)z * sA; B += (long)z * sB; C += (long)z * sC;
  const int m0 = blockIdx.y * 64, n0 = blockIdx.x * 64;
  const int tid = threadIdx.x, lane = tid & 31, w = tid >> 5;
  const int wr = (w >> 1) * 32, wc = (w & 1) * 32;
  v8f acc00 = {}, acc01 = {}, acc10 = {}, acc11 = {};
  const int nk = K >> 6;

#if HAS_TDM
  if (w == 0) {
    tdm_load_tile(lds_offset(&As[0][0]), A + (long)m0 * lda, lda, 64, 64, 4);
    if (!TRANSB) tdm_load_tile(lds_offset(&Bs[0][0]), B + n0, ldb, 64, 64, 4);
  }
#endif
  for (int kt = 0; kt < nk; ++kt) {
    const int k0 = kt << 6;
    const int cur = kt & 1, nxt = cur ^ 1;
#if HAS_TDM
    if (TRANSB) stage_tile_transposed(Bs[cur], B + (long)n0 * ldb + k0, ldb, tid);
    if (w == 0) {
      if (kt + 1 < nk) {
        tdm_load_tile(lds_offset(&As[nxt][0]), A + (long)m0 * lda + k0 + 64, lda, 64, 64, 4);
        if (!TRANSB) {
          tdm_load_tile(lds_offset(&Bs[nxt][0]), B + (long)(k0 + 64) * ldb + n0, ldb, 64, 64, 4);
          __builtin_amdgcn_s_wait_tensorcnt(2);
        } else {
          __builtin_amdgcn_s_wait_tensorcnt(1);
        }
      } else {
        __builtin_amdgcn_s_wait_tensorcnt(0);
      }
    }
#else
    stage_tile_rowmajor(As[cur], A + (long)m0 * lda + k0, lda, tid);
    if (!TRANSB) stage_tile_rowmajor(Bs[cur], B + (long)k0 * ldb + n0, ldb, tid);
    else         stage_tile_transposed(Bs[cur], B + (long)n0 * ldb + k0, ldb, tid);
#endif
    __syncthreads();
    #pragma unroll
    for (int kk = 0; kk < 64; kk += 32) {
      v16bf a0 = frag_a(As[cur], LDK, wr,      kk, lane);
      v16bf a1 = frag_a(As[cur], LDK, wr + 16, kk, lane);
      v16bf b0 = frag_b(Bs[cur], LDK, kk, wc,      lane);
      v16bf b1 = frag_b(Bs[cur], LDK, kk, wc + 16, lane);
      acc00 = wmma_bf16(a0, b0, acc00);
      acc01 = wmma_bf16(a0, b1, acc01);
      acc10 = wmma_bf16(a1, b0, acc10);
      acc11 = wmma_bf16(a1, b1, acc11);
    }
    __syncthreads();
  }

  const int lr = lane & 15, hi = lane >> 4;
  #pragma unroll
  for (int vg = 0; vg < 8; ++vg) {
    int mA = m0 + wr + vg + 8 * hi, mB = mA + 16;
    int nA = n0 + wc + lr,          nB = nA + 16;
    if (ACCUM) {
      C[(long)mA * ldc + nA] += acc00[vg];
      C[(long)mA * ldc + nB] += acc01[vg];
      C[(long)mB * ldc + nA] += acc10[vg];
      C[(long)mB * ldc + nB] += acc11[vg];
    } else {
      C[(long)mA * ldc + nA] = acc00[vg];
      C[(long)mA * ldc + nB] = acc01[vg];
      C[(long)mB * ldc + nA] = acc10[vg];
      C[(long)mB * ldc + nB] = acc11[vg];
    }
  }
}

// ---------------------------------------------------------------------------
// 128x128-tile GEMM core (8 waves, each 32x64 = 2x4 WMMA tiles).
// Dynamic LDS: 2 A tiles (128x64) + 2 B tiles (64x128), ~70 KB.
// Buffer addresses computed by offset arithmetic (no pointer arrays: a
// brace-initialized pointer array from extern __shared__ emits an
// addrspacecast static initializer that ld.lld rejects).
// ---------------------------------------------------------------------------
#define SM128_BYTES ((2 * ATILE128 + 2 * BTILE128) * (int)sizeof(unsigned short))

__device__ __forceinline__ void gemm128_core(
    const unsigned short* __restrict__ A, const unsigned short* __restrict__ B,
    int K, int lda, int ldb, int m0, int n0,
    int tid, int lane, int w, v8f acc[2][4])
{
  extern __shared__ unsigned short sm16[];
  unsigned short* const Abase = sm16;                  // [2][ATILE128]
  unsigned short* const Bbase = sm16 + 2 * ATILE128;   // [2][BTILE128]
  const int wr = (w >> 1) * 32, wcol = (w & 1) * 64;
  const int nk = K >> 6;

#if HAS_TDM
  if (w == 0) {
    tdm_load_tile(lds_offset(Abase), A + (long)m0 * lda, lda, 128, 64, 4);
    tdm_load_tile(lds_offset(Bbase), B + n0, ldb, 64, 128, 5);
  }
#endif
  for (int kt = 0; kt < nk; ++kt) {
    const int k0 = kt << 6;
    const int cur = kt & 1, nxt = cur ^ 1;
    unsigned short* Acur = Abase + cur * ATILE128;
    unsigned short* Bcur = Bbase + cur * BTILE128;
#if HAS_TDM
    if (w == 0) {
      if (kt + 1 < nk) {
        tdm_load_tile(lds_offset(Abase + nxt * ATILE128), A + (long)m0 * lda + k0 + 64, lda, 128, 64, 4);
        tdm_load_tile(lds_offset(Bbase + nxt * BTILE128), B + (long)(k0 + 64) * ldb + n0, ldb, 64, 128, 5);
        __builtin_amdgcn_s_wait_tensorcnt(2);
      } else {
        __builtin_amdgcn_s_wait_tensorcnt(0);
      }
    }
#else
    stage_a128(Acur, A + (long)m0 * lda + k0, lda, tid);
    stage_b128(Bcur, B + (long)k0 * ldb + n0, ldb, tid);
#endif
    __syncthreads();
    #pragma unroll
    for (int kk = 0; kk < 64; kk += 32) {
      v16bf a0 = frag_a(Acur, LDK, wr,      kk, lane);
      v16bf a1 = frag_a(Acur, LDK, wr + 16, kk, lane);
      #pragma unroll
      for (int tj = 0; tj < 4; ++tj) {
        v16bf bt = frag_b(Bcur, LDB128, kk, wcol + tj * 16, lane);
        acc[0][tj] = wmma_bf16(a0, bt, acc[0][tj]);
        acc[1][tj] = wmma_bf16(a1, bt, acc[1][tj]);
      }
    }
    __syncthreads();
  }
}

// 128x128 GEMM, plain store (used for the final projection).
__global__ __launch_bounds__(256) void gemm_bf16_128(
    const unsigned short* __restrict__ A, const unsigned short* __restrict__ B,
    float* __restrict__ C, int K, int lda, int ldb, int ldc)
{
  const int m0 = blockIdx.y * 128, n0 = blockIdx.x * 128;
  const int tid = threadIdx.x, lane = tid & 31, w = tid >> 5;
  const int wr = (w >> 1) * 32, wcol = (w & 1) * 64;
  v8f acc[2][4] = {};
  gemm128_core(A, B, K, lda, ldb, m0, n0, tid, lane, w, acc);

  const int lr = lane & 15, hi = lane >> 4;
  #pragma unroll
  for (int ti = 0; ti < 2; ++ti)
    #pragma unroll
    for (int tj = 0; tj < 4; ++tj)
      #pragma unroll
      for (int vg = 0; vg < 8; ++vg) {
        int m = m0 + wr + ti * 16 + vg + 8 * hi;
        int n = n0 + wcol + tj * 16 + lr;
        C[(long)m * ldc + n] = acc[ti][tj][vg];
      }
}

// ---------------------------------------------------------------------------
// QKV GEMM (128x128 tiles): cxb[49152x512] @ wqkvb[512x1536]; epilogue splits
// heads into bf16 q (scaled, first 4096 seq rows), k, v.
// ---------------------------------------------------------------------------
__device__ __forceinline__ void qkv_scatter(int m, int n, float val,
    unsigned short* __restrict__ qf, unsigned short* __restrict__ kf,
    unsigned short* __restrict__ vf) {
  int bb = m / 12288;
  int seq = m - bb * 12288;
  int part = n >> 9;
  int cc = n & 511;
  int h = cc >> 6, d = cc & 63;
  long bh = (long)bb * 8 + h;
  if (part == 0) {
    if (seq < 4096) qf[(bh * 4096 + seq) * 64 + d] = f2bf(val * 0.125f);
  } else if (part == 1) {
    kf[(bh * 12288 + seq) * 64 + d] = f2bf(val);
  } else {
    vf[(bh * 12288 + seq) * 64 + d] = f2bf(val);
  }
}

__global__ __launch_bounds__(256) void qkv_gemm(
    const unsigned short* __restrict__ A, const unsigned short* __restrict__ B,
    unsigned short* __restrict__ qf, unsigned short* __restrict__ kf,
    unsigned short* __restrict__ vf)
{
  const int m0 = blockIdx.y * 128, n0 = blockIdx.x * 128;
  const int tid = threadIdx.x, lane = tid & 31, w = tid >> 5;
  const int wr = (w >> 1) * 32, wcol = (w & 1) * 64;
  v8f acc[2][4] = {};
  gemm128_core(A, B, 512, 512, 1536, m0, n0, tid, lane, w, acc);

  const int lr = lane & 15, hi = lane >> 4;
  #pragma unroll
  for (int ti = 0; ti < 2; ++ti)
    #pragma unroll
    for (int tj = 0; tj < 4; ++tj)
      #pragma unroll
      for (int vg = 0; vg < 8; ++vg) {
        int m = m0 + wr + ti * 16 + vg + 8 * hi;
        int n = n0 + wcol + tj * 16 + lr;
        qkv_scatter(m, n, acc[ti][tj][vg], qf, kf, vf);
      }
}

// ---------------------------------------------------------------------------
// Elementwise converters
// ---------------------------------------------------------------------------
__global__ void cvt_f32_bf16(const float* __restrict__ src, unsigned short* __restrict__ dst, long n)
{
  long idx = (long)blockIdx.x * blockDim.x + threadIdx.x;
  if (idx < n) dst[idx] = f2bf(src[idx]);
}

__global__ void concat_cx(const float* __restrict__ x, const float* __restrict__ ps,
                          unsigned short* __restrict__ dst)
{
  long idx = (long)blockIdx.x * blockDim.x + threadIdx.x;
  if (idx >= (long)49152 * 512) return;
  int c = (int)(idx & 511);
  int row = (int)(idx >> 9);
  int bb = row / 12288;
  int seq = row - bb * 12288;
  const float* src = (seq < 4096)
      ? (x  + ((long)bb * 4096 + seq) * 512 + c)
      : (ps + ((long)bb * 8192 + (seq - 4096)) * 512 + c);
  dst[idx] = f2bf(*src);
}

// ---------------------------------------------------------------------------
// Landmarks: group means along the sequence dim (bf16 in/out, f32 accum).
// ---------------------------------------------------------------------------
__global__ void group_mean(const unsigned short* __restrict__ src, unsigned short* __restrict__ dst,
                           int nrows, int g)
{
  int idx = blockIdx.x * blockDim.x + threadIdx.x;
  if (idx >= 32 * 64 * 64) return;
  int d = idx & 63, j = (idx >> 6) & 63, bh = idx >> 12;
  const unsigned short* p = src + ((long)bh * nrows + (long)j * g) * 64 + d;
  float a = 0.0f;
  for (int r = 0; r < g; ++r) a += bf2f(p[(long)r * 64]);
  dst[((long)bh * 64 + j) * 64 + d] = f2bf(a / (float)g);
}

// ---------------------------------------------------------------------------
// Row softmax (wave32 per row); writes f32 in place and optional bf16 copy.
// ---------------------------------------------------------------------------
__global__ __launch_bounds__(256) void softmax_rows(float* __restrict__ p,
                                                    unsigned short* __restrict__ pb,
                                                    int R, int Cn)
{
  int wave = (blockIdx.x * 256 + threadIdx.x) >> 5;
  int lane = threadIdx.x & 31;
  if (wave >= R) return;
  float* row = p + (long)wave * Cn;
  float mx = -3.4e38f;
  for (int c = lane; c < Cn; c += 32) mx = fmaxf(mx, row[c]);
  #pragma unroll
  for (int st = 16; st > 0; st >>= 1) mx = fmaxf(mx, __shfl_xor(mx, st, 32));
  float sum = 0.0f;
  for (int c = lane; c < Cn; c += 32) sum += __expf(row[c] - mx);
  #pragma unroll
  for (int st = 16; st > 0; st >>= 1) sum += __shfl_xor(sum, st, 32);
  float inv = 1.0f / sum;
  if (pb) {
    unsigned short* rb = pb + (long)wave * Cn;
    for (int c = lane; c < Cn; c += 32) {
      float v = __expf(row[c] - mx) * inv;
      row[c] = v; rb[c] = f2bf(v);
    }
  } else {
    for (int c = lane; c < Cn; c += 32) row[c] = __expf(row[c] - mx) * inv;
  }
}

// ---------------------------------------------------------------------------
// Global scale for pinv init: s = max(row-abs-sums) * max(col-abs-sums).
// ---------------------------------------------------------------------------
__global__ __launch_bounds__(256) void pinv_scale_kernel(
    const float* __restrict__ attn2, float* __restrict__ s)
{
  __shared__ float red[256];
  float rmax = 0.0f, cmax = 0.0f;
  for (int idx = threadIdx.x; idx < 2048; idx += 256) {
    int bh = idx >> 6, r = idx & 63;
    const float* p = attn2 + (long)bh * 4096 + r * 64;
    float a = 0.0f;
    for (int j = 0; j < 64; ++j) a += fabsf(p[j]);
    rmax = fmaxf(rmax, a);
    const float* q = attn2 + (long)bh * 4096 + r;
    float b = 0.0f;
    for (int i = 0; i < 64; ++i) b += fabsf(q[i * 64]);
    cmax = fmaxf(cmax, b);
  }
  red[threadIdx.x] = rmax; __syncthreads();
  for (int st = 128; st > 0; st >>= 1) {
    if (threadIdx.x < st) red[threadIdx.x] = fmaxf(red[threadIdx.x], red[threadIdx.x + st]);
    __syncthreads();
  }
  float rm = red[0]; __syncthreads();
  red[threadIdx.x] = cmax; __syncthreads();
  for (int st = 128; st > 0; st >>= 1) {
    if (threadIdx.x < st) red[threadIdx.x] = fmaxf(red[threadIdx.x], red[threadIdx.x + st]);
    __syncthreads();
  }
  if (threadIdx.x == 0) s[0] = rm * red[0];
}

// ---------------------------------------------------------------------------
// In-LDS 64x64 WMMA matmuls for the Newton-Schulz pinv (f32 LDS, stride 65).
// ---------------------------------------------------------------------------
#define LDF 65

__device__ __forceinline__ v16bf frag_a_f32(const float* A, int row_base, int k0, int lane) {
  int lr = lane & 15, hi = lane >> 4;
  const float* p = A + (row_base + lr) * LDF + k0 + 8 * hi;
  FragBF f;
  #pragma unroll
  for (int j = 0; j < 8; ++j) { f.u[j] = f2bf(p[j]); f.u[8 + j] = f2bf(p[16 + j]); }
  return f.v;
}
__device__ __forceinline__ v16bf frag_b_f32(const float* B, int k0, int col_base, int lane) {
  int lr = lane & 15, hi = lane >> 4;
  const float* p = B + (k0 + 8 * hi) * LDF + col_base + lr;
  FragBF f;
  #pragma unroll
  for (int j = 0; j < 8; ++j) { f.u[j] = f2bf(p[j * LDF]); f.u[8 + j] = f2bf(p[(16 + j) * LDF]); }
  return f.v;
}

__device__ __forceinline__ void mm64(float* D, const float* A, const float* B, int lane, int w) {
  const int wr = (w >> 1) * 32, wc = (w & 1) * 32;
  v8f a00 = {}, a01 = {}, a10 = {}, a11 = {};
  #pragma unroll
  for (int kk = 0; kk < 64; kk += 32) {
    v16bf fa0 = frag_a_f32(A, wr,      kk, lane);
    v16bf fa1 = frag_a_f32(A, wr + 16, kk, lane);
    v16bf fb0 = frag_b_f32(B, kk, wc,      lane);
    v16bf fb1 = frag_b_f32(B, kk, wc + 16, lane);
    a00 = wmma_bf16(fa0, fb0, a00);
    a01 = wmma_bf16(fa0, fb1, a01);
    a10 = wmma_bf16(fa1, fb0, a10);
    a11 = wmma_bf16(fa1, fb1, a11);
  }
  __syncthreads();
  int lr = lane & 15, hi = lane >> 4;
  #pragma unroll
  for (int vg = 0; vg < 8; ++vg) {
    int mA = wr + vg + 8 * hi, nA = wc + lr;
    D[mA * LDF + nA]             = a00[vg];
    D[mA * LDF + nA + 16]        = a01[vg];
    D[(mA + 16) * LDF + nA]      = a10[vg];
    D[(mA + 16) * LDF + nA + 16] = a11[vg];
  }
  __syncthreads();
}

__device__ __forceinline__ void ew_ci_minus(float* dst, const float* src, float c, int tid) {
  for (int idx = tid; idx < 4096; idx += 128) {
    int i = idx >> 6, j = idx & 63;
    dst[i * LDF + j] = ((i == j) ? c : 0.0f) - src[i * LDF + j];
  }
  __syncthreads();
}

__global__ __launch_bounds__(128) void pinv_kernel(
    const float* __restrict__ attn2, const float* __restrict__ t3v,
    const float* __restrict__ sden, unsigned short* __restrict__ t2)
{
  extern __shared__ float sm[];
  float* X  = sm;
  float* Zp = sm + 1 * 64 * LDF;
  float* T  = sm + 2 * 64 * LDF;
  float* Ua = sm + 3 * 64 * LDF;
  float* Ub = sm + 4 * 64 * LDF;
  const int bh = blockIdx.x;
  const int tid = threadIdx.x, lane = tid & 31, w = tid >> 5;
  const float inv_s = 1.0f / sden[0];
  const float* Xg = attn2 + (long)bh * 4096;
  for (int idx = tid; idx < 4096; idx += 128) {
    int i = idx >> 6, j = idx & 63;
    float v = Xg[idx];
    X[i * LDF + j]  = v;
    Zp[j * LDF + i] = v * inv_s;       // z0 = x^T / (max(col)*max(row))
  }
  __syncthreads();

  for (int it = 0; it < 6; ++it) {
    mm64(T, X, Zp, lane, w);           // T  = x @ z
    ew_ci_minus(Ua, T, 7.0f, tid);     // Ua = 7I - T
    mm64(Ub, T, Ua, lane, w);          // Ub = T @ Ua
    ew_ci_minus(Ua, Ub, 15.0f, tid);   // Ua = 15I - Ub
    mm64(Ub, T, Ua, lane, w);          // Ub = T @ Ua
    ew_ci_minus(Ua, Ub, 13.0f, tid);   // Ua = 13I - Ub
    mm64(Ub, Zp, Ua, lane, w);         // Ub = z @ Ua
    for (int idx = tid; idx < 4096; idx += 128) {
      int i = idx >> 6, j = idx & 63;
      Zp[i * LDF + j] = 0.25f * Ub[i * LDF + j];
    }
    __syncthreads();
  }

  const float* Tg = t3v + (long)bh * 4096;
  for (int idx = tid; idx < 4096; idx += 128) {
    int i = idx >> 6, j = idx & 63;
    T[i * LDF + j] = Tg[idx];
  }
  __syncthreads();
  mm64(Ub, Zp, T, lane, w);
  unsigned short* out = t2 + (long)bh * 4096;
  for (int idx = tid; idx < 4096; idx += 128) {
    int i = idx >> 6, j = idx & 63;
    out[idx] = f2bf(Ub[i * LDF + j]);
  }
}

// ---------------------------------------------------------------------------
// Depthwise residual conv (33 taps over seq), bf16 v -> f32 outh.
// ---------------------------------------------------------------------------
__global__ void conv_res(const unsigned short* __restrict__ vf, const float* __restrict__ wconv,
                         float* __restrict__ outh)
{
  long idx = (long)blockIdx.x * blockDim.x + threadIdx.x;
  if (idx >= (long)32 * 4096 * 64) return;
  int d  = (int)(idx & 63);
  int i  = (int)((idx >> 6) & 4095);
  int bh = (int)(idx >> 18);
  int h  = bh & 7;
  const unsigned short* v = vf + (long)bh * 12288 * 64 + d;
  float a = 0.0f;
  #pragma unroll
  for (int j = 0; j < 33; ++j) {
    int s = i + j - 16;
    if (s >= 0 && s < 12288) a += wconv[h * 33 + j] * bf2f(v[(long)s * 64]);
  }
  outh[idx] = a;
}

__global__ void transpose_heads(const float* __restrict__ outh, unsigned short* __restrict__ outt)
{
  long idx = (long)blockIdx.x * blockDim.x + threadIdx.x;
  if (idx >= (long)4 * 4096 * 512) return;
  int c = (int)(idx & 511);
  int i = (int)((idx >> 9) & 4095);
  int b = (int)(idx >> 21);
  int h = c >> 6, d = c & 63;
  outt[idx] = f2bf(outh[(((long)(b * 8 + h)) * 4096 + i) * 64 + d]);
}

// ---------------------------------------------------------------------------
// Launcher
// ---------------------------------------------------------------------------
extern "C" void kernel_launch(void* const* d_in, const int* in_sizes, int n_in,
                              void* d_out, int out_size, void* d_ws, size_t ws_size,
                              hipStream_t stream)
{
  const float* x    = (const float*)d_in[0];  // [4,4096,512]
  const float* ps   = (const float*)d_in[1];  // [4,8192,512]
  const float* wqkv = (const float*)d_in[2];  // [512,1536]
  const float* wout = (const float*)d_in[3];  // [512,512]
  const float* wcv  = (const float*)d_in[4];  // [8,1,33,1]
  float* out = (float*)d_out;                 // [4,4096,512]

  // f32 region
  float* wf = (float*)d_ws;
  float* sim1 = wf; wf += 8388608;    // [32,4096,64]  -> attn1
  float* sim2 = wf; wf += 131072;     // [32,64,64]    -> attn2
  float* sim3 = wf; wf += 25165824;   // [32,64,12288] -> attn3
  float* t3v  = wf; wf += 131072;     // [32,64,64]
  float* outh = wf; wf += 8388608;    // [32,4096,64]
  float* sden = wf; wf += 128;        // scalar
  // bf16 region
  unsigned short* wb = (unsigned short*)wf;
  unsigned short* cxb    = wb; wb += 25165824;  // [49152,512]
  unsigned short* wqkvb  = wb; wb += 786432;    // [512,1536]
  unsigned short* woutb  = wb; wb += 262144;    // [512,512]
  unsigned short* qfb    = wb; wb += 8388608;   // [32,4096,64] (scaled)
  unsigned short* kfb    = wb; wb += 25165824;  // [32,12288,64]
  unsigned short* vfb    = wb; wb += 25165824;  // [32,12288,64]
  unsigned short* qlb    = wb; wb += 131072;    // [32,64,64]
  unsigned short* klb    = wb; wb += 131072;    // [32,64,64]
  unsigned short* attn1b = wb; wb += 8388608;   // [32,4096,64]
  unsigned short* attn3b = wb; wb += 25165824;  // [32,64,12288]
  unsigned short* t2b    = wb; wb += 131072;    // [32,64,64]
  unsigned short* outtb  = wb; wb += 8388608;   // [4,4096,512]

  dim3 blk(128);

  // 0) one-time bf16 conversions (producers)
  concat_cx<<<(49152L * 512) / 256, 256, 0, stream>>>(x, ps, cxb);
  cvt_f32_bf16<<<786432 / 256, 256, 0, stream>>>(wqkv, wqkvb, 786432);
  cvt_f32_bf16<<<262144 / 256, 256, 0, stream>>>(wout, woutb, 262144);

  // 1) fused qkv GEMM + head split (128x128 tiles)
  qkv_gemm<<<dim3(1536 / 128, 49152 / 128, 1), 256, SM128_BYTES, stream>>>(
      cxb, wqkvb, qfb, kfb, vfb);

  // 2) landmarks
  group_mean<<<131072 / 256, 256, 0, stream>>>(qfb, qlb, 4096, 64);
  group_mean<<<131072 / 256, 256, 0, stream>>>(kfb, klb, 12288, 192);

  // 3) similarity GEMMs (B transposed: stored [N][K])
  gemm_bf16<true, false><<<dim3(1, 64, 32), blk, 0, stream>>>(
      qfb, klb, sim1, 4096, 64, 64, 64, 64, 64, 4096L * 64, 64L * 64, 4096L * 64);
  gemm_bf16<true, false><<<dim3(1, 1, 32), blk, 0, stream>>>(
      qlb, klb, sim2, 64, 64, 64, 64, 64, 64, 64L * 64, 64L * 64, 64L * 64);
  gemm_bf16<true, false><<<dim3(192, 1, 32), blk, 0, stream>>>(
      qlb, kfb, sim3, 64, 12288, 64, 64, 64, 12288, 64L * 64, 12288L * 64, 64L * 12288);

  // 4) softmaxes (bf16 copies for downstream GEMM operands)
  softmax_rows<<<131072 / 8, 256, 0, stream>>>(sim1, attn1b, 131072, 64);
  softmax_rows<<<2048 / 8, 256, 0, stream>>>(sim2, (unsigned short*)nullptr, 2048, 64);
  softmax_rows<<<2048 / 8, 256, 0, stream>>>(sim3, attn3b, 2048, 12288);

  // 5) pinv scale + t3v = attn3 @ v
  pinv_scale_kernel<<<1, 256, 0, stream>>>(sim2, sden);
  gemm_bf16<false, false><<<dim3(1, 1, 32), blk, 0, stream>>>(
      attn3b, vfb, t3v, 64, 64, 12288, 12288, 64, 64, 64L * 12288, 12288L * 64, 64L * 64);

  // 6) conv residual into outh
  conv_res<<<8388608 / 256, 256, 0, stream>>>(vfb, wcv, outh);

  // 7) Newton-Schulz pinv + t2 = z @ t3v
  pinv_kernel<<<32, 128, 5 * 64 * LDF * sizeof(float), stream>>>(sim2, t3v, sden, t2b);

  // 8) outh += attn1 @ t2
  gemm_bf16<false, true><<<dim3(1, 64, 32), blk, 0, stream>>>(
      attn1b, t2b, outh, 4096, 64, 64, 64, 64, 64, 4096L * 64, 64L * 64, 4096L * 64);

  // 9) merge heads, final projection (128x128 tiles)
  transpose_heads<<<8388608 / 256, 256, 0, stream>>>(outh, outtb);
  gemm_bf16_128<<<dim3(512 / 128, 16384 / 128, 1), 256, SM128_BYTES, stream>>>(
      outtb, woutb, out, 512, 512, 512, 512);
}
